// SelfAttention_2190433321310
// MI455X (gfx1250) — matrix operand order repaired and checked by reading, not compiled, not run on hardware
//
#include <hip/hip_runtime.h>
#include <hip/hip_bf16.h>

// ---------------------------------------------------------------------------
// Self-attention forward for MI455X (gfx1250), bf16 WMMA everywhere.
// B=4, S=2048, D=DK=DV=1024.  All matmul via v_wmma_f32_16x16x32_bf16.
//
// MI455X strategy: 192MB L2 + 23.3TB/s HBM -> L2-centric GEMMs with NO LDS
// staging; all WMMA fragments are per-lane contiguous 16B/32B loads
// (global_load_b128 / ds_load_b128).  Weights and V are pre-transposed so
// every B-operand read is contiguous.  Flash attention keeps only q and the
// probability tile in LDS (the only truly shared data).  bf16 conversion is
// done with native hardware converts (v_cvt_*bf16_f32), not bit twiddling.
// ---------------------------------------------------------------------------

#define B_DIM 4
#define S_DIM 2048
#define D_DIM 1024
#define M_DIM (B_DIM * S_DIM)   // 8192 token rows

typedef __attribute__((ext_vector_type(16))) __bf16 bfrag;   // 8 VGPRs: 16x32 A / 32x16 B
typedef __attribute__((ext_vector_type(8)))  float  v8f;     // 8 VGPRs: 16x16 f32 C/D
typedef __attribute__((ext_vector_type(2)))  __bf16 bf16x2;
typedef __attribute__((ext_vector_type(2)))  float  f32x2;

// ---- converters (native HW convert, RTE) ----------------------------------
__device__ __forceinline__ __bf16 f32_to_bf16(float f) { return (__bf16)f; }

__device__ __forceinline__ bf16x2 f32x2_to_bf16x2(float a, float b) {
    f32x2 v = {a, b};
    return __builtin_convertvector(v, bf16x2);   // -> packed bf16 convert
}

// ---- WMMA fragment loaders (contiguous-k layout, wave32) -------------------
// A matrix 16x32 bf16, row-major source with leading dim ld:
//   M = lane%16;  lanes 0-15 hold K 0-15, lanes 16-31 hold K 16-31.
// Per lane: one contiguous 32B run -> 2x b128 loads.
__device__ __forceinline__ bfrag load_a_frag(const __bf16* p, int ld, int lane) {
    int lm = lane & 15;
    int kh = (lane & 16) ? 16 : 0;
    const __bf16* row = p + (long)lm * ld + kh;
    bfrag a;
#pragma unroll
    for (int e = 0; e < 8; ++e) a[e] = row[e];
#pragma unroll
    for (int e = 0; e < 8; ++e) a[8 + e] = row[8 + e];
    return a;
}

// B matrix 32x16 bf16 where memory is [N][K] (transposed source):
//   N = lane%16; lanes 0-15 hold K 0-15, lanes 16-31 hold K 16-31.
// Per lane: one contiguous 32B run -> 2x b128 loads.
__device__ __forceinline__ bfrag load_b_frag_t(const __bf16* p, int ld, int lane) {
    int n  = lane & 15;
    int kh = (lane & 16) ? 16 : 0;
    const __bf16* row = p + (long)n * ld + kh;
    bfrag b;
#pragma unroll
    for (int e = 0; e < 16; ++e) b[e] = row[e];
    return b;
}

__device__ __forceinline__ v8f wmma_bf16(bfrag a, bfrag b, v8f c) {
    return __builtin_amdgcn_wmma_f32_16x16x32_bf16(false, a, false, b, (short)0, c,
                                                   false, false);
}

// ---------------------------------------------------------------------------
// fp32 -> bf16 conversion (row-major, 4 elems/thread) -- used for x
// ---------------------------------------------------------------------------
__global__ __launch_bounds__(256) void cvt_f32_bf16(const float* __restrict__ src,
                                                    __bf16* __restrict__ dst, int n) {
    int i = (blockIdx.x * 256 + threadIdx.x) * 4;
    if (i < n) {
        float4 f = *(const float4*)(src + i);
        __attribute__((aligned(8))) bf16x2 o[2];
        o[0] = f32x2_to_bf16x2(f.x, f.y);
        o[1] = f32x2_to_bf16x2(f.z, f.w);
        *(uint2*)(dst + i) = *(uint2*)o;
    }
}

// ---------------------------------------------------------------------------
// fp32 [K][N] -> bf16 transposed [N][K]  -- used for weights (2MB, L2-resident)
// ---------------------------------------------------------------------------
__global__ __launch_bounds__(256) void cvt_transpose_f32_bf16(const float* __restrict__ src,
                                                              __bf16* __restrict__ dst,
                                                              int K, int N) {
    int i = blockIdx.x * 256 + threadIdx.x;   // coalesced read along N
    if (i < K * N) {
        int k = i / N, n = i - k * N;
        dst[(size_t)n * K + k] = f32_to_bf16(src[i]);
    }
}

// ---------------------------------------------------------------------------
// bf16 GEMM: C[M,N] = A[M,K] * B[K,N] with B supplied TRANSPOSED as BT[N][K].
// No LDS: all fragments are contiguous per-lane global b128 loads; reuse is
// served by the 192MB L2.  Block tile 128x128 (8 waves, 4x2), wave 32x64.
// OUT_MODE: 0 = bf16 row-major, 1 = bf16 transposed-per-batch [b][n][s]
//           (vectorized uint4 stores thanks to the C layout), 2 = f32 row-major.
// ---------------------------------------------------------------------------
template <int OUT_MODE>
__global__ __launch_bounds__(256) void gemm_bf16_bt(const __bf16* __restrict__ A,
                                                    const __bf16* __restrict__ BT,
                                                    void* __restrict__ C,
                                                    int M, int N, int K) {
    int tid  = threadIdx.x;
    int lane = tid & 31;
    int w    = tid >> 5;
    int wm   = (w & 3) * 32;   // wave row offset in tile
    int wn   = (w >> 2) * 64;  // wave col offset in tile
    int rowBase = blockIdx.y * 128;
    int colBase = blockIdx.x * 128;

    const __bf16* Arow0 = A  + (size_t)(rowBase + wm) * K;
    const __bf16* Brow0 = BT + (size_t)(colBase + wn) * K;

    v8f zero = {0.f, 0.f, 0.f, 0.f, 0.f, 0.f, 0.f, 0.f};
    v8f acc[2][4];
#pragma unroll
    for (int mt = 0; mt < 2; ++mt)
#pragma unroll
        for (int nt = 0; nt < 4; ++nt) acc[mt][nt] = zero;

    for (int kt = 0; kt < K; kt += 32) {
        if (kt + 32 < K) {  // -> global_prefetch_b8
            __builtin_prefetch(Arow0 + (size_t)lane * K + kt + 32, 0, 1);
            __builtin_prefetch(Brow0 + (size_t)lane * K + kt + 32, 0, 1);
        }
        bfrag af[2];
#pragma unroll
        for (int mt = 0; mt < 2; ++mt)
            af[mt] = load_a_frag(Arow0 + (size_t)(mt * 16) * K + kt, K, lane);
        bfrag bf[4];
#pragma unroll
        for (int nt = 0; nt < 4; ++nt)
            bf[nt] = load_b_frag_t(Brow0 + (size_t)(nt * 16) * K + kt, K, lane);
#pragma unroll
        for (int mt = 0; mt < 2; ++mt)
#pragma unroll
            for (int nt = 0; nt < 4; ++nt)
                acc[mt][nt] = wmma_bf16(af[mt], bf[nt], acc[mt][nt]);
    }

    // C/D layout: row = vgpr + 8*(lane/16), col = lane%16
    int cl = lane & 15;
    int rh = (lane & 16) ? 8 : 0;
#pragma unroll
    for (int mt = 0; mt < 2; ++mt)
#pragma unroll
        for (int nt = 0; nt < 4; ++nt) {
            int gcol = colBase + wn + nt * 16 + cl;
            if (OUT_MODE == 1) {
                // per lane: 8 consecutive token rows of one column -> one uint4
                int grow0 = rowBase + wm + mt * 16 + rh;      // multiple of 8
                int b  = grow0 >> 11;                         // / S_DIM
                int s0 = grow0 & (S_DIM - 1);
                __attribute__((aligned(16))) bf16x2 tmp[4];
#pragma unroll
                for (int r = 0; r < 4; ++r)
                    tmp[r] = f32x2_to_bf16x2(acc[mt][nt][2 * r], acc[mt][nt][2 * r + 1]);
                __bf16* dst = (__bf16*)C +
                    ((size_t)b * D_DIM + gcol) * S_DIM + s0;
                *(uint4*)dst = *(uint4*)tmp;
            } else {
#pragma unroll
                for (int r = 0; r < 8; ++r) {
                    size_t grow = (size_t)(rowBase + wm + mt * 16 + r + rh);
                    if (OUT_MODE == 0)
                        ((__bf16*)C)[grow * N + gcol] = f32_to_bf16(acc[mt][nt][r]);
                    else
                        ((float*)C)[grow * N + gcol] = acc[mt][nt][r];
                }
            }
        }
}

// ---------------------------------------------------------------------------
// Flash attention: one workgroup per (batch, 16-row q block), 8 waves.
//  - q block (16x1024 bf16) in LDS (shared by all waves, all kv blocks)
//  - k B-fragments read directly from global row-major k (contiguous runs)
//  - v B-fragments read directly from global TRANSPOSED vT[b][d][s]
//  - cooperative online softmax via LDS (m, l, probabilities)
//  - wave w: score tile kv rows [w*16,+16); O accum V cols [w*128,+128)
// Static LDS ~47KB.
// ---------------------------------------------------------------------------
#define QLD 1032   // 1024 + pad (pitch 2064B, 16B-aligned)
#define PLD 136    // 128 + pad  (pitch 272B,  16B-aligned)

__global__ __launch_bounds__(256) void flash_attn(const __bf16* __restrict__ Q,
                                                  const __bf16* __restrict__ Kmat,
                                                  const __bf16* __restrict__ VT,
                                                  __bf16* __restrict__ O) {
    __shared__ __bf16 qT[16 * QLD];    // 33024 B
    __shared__ float  pT[16 * PLD];    //  8704 B
    __shared__ __bf16 pB[16 * PLD];    //  4352 B
    __shared__ float  red[304];        //  1216 B
    float* mR   = red;          // [16] running max
    float* lR   = red + 16;     // [16] running sum
    float* fac  = red + 32;     // [16] rescale factor
    float* pmax = red + 48;     // [16][8]
    float* psum = red + 176;    // [16][8]

    const int tid  = threadIdx.x;
    const int lane = tid & 31;
    const int w    = tid >> 5;
    const int cl   = lane & 15;
    const int rh   = (lane & 16) ? 8 : 0;

    const int qb    = blockIdx.x;
    const int bidx  = blockIdx.y;
    const int qRow0 = qb * 16;

    const __bf16* qG = Q    + ((size_t)bidx * S_DIM + qRow0) * D_DIM;
    const __bf16* kG = Kmat + (size_t)bidx * S_DIM * D_DIM;
    const __bf16* vG = VT   + (size_t)bidx * D_DIM * S_DIM;   // [d][s]

    if (tid < 16) { mR[tid] = -3.0e38f; lR[tid] = 0.0f; }

    // stage q block: 16x1024 bf16 = 2048 uint4, 8 per thread
#pragma unroll
    for (int j = 0; j < 8; ++j) {
        int u = tid + j * 256;
        int r = u >> 7;
        int c = (u & 127) * 8;
        *(uint4*)((char*)qT + r * (QLD * 2) + c * 2) =
            *(const uint4*)(qG + (size_t)r * D_DIM + c);
    }
    __syncthreads();

    v8f zero = {0.f, 0.f, 0.f, 0.f, 0.f, 0.f, 0.f, 0.f};
    v8f acc[8];
#pragma unroll
    for (int nt = 0; nt < 8; ++nt) acc[nt] = zero;

    const float kscale  = 0.03125f;                  // 1/sqrt(1024)
    const int   nBlocks = (qRow0 + 16 + 127) >> 7;   // causal early exit

    for (int kvb = 0; kvb < nBlocks; ++kvb) {
        const int kvBase = kvb << 7;

        // ---- scores: S_w(16x16) = q(16xDK) . k[w*16..+16]^T
        const __bf16* kRow = kG + (size_t)(kvBase + w * 16) * D_DIM;
        v8f sc = zero;
#pragma unroll 4
        for (int ks = 0; ks < D_DIM / 32; ++ks) {
            bfrag a  = load_a_frag(qT + ks * 32, QLD, lane);
            bfrag bb = load_b_frag_t(kRow + ks * 32, D_DIM, lane);
            sc = wmma_bf16(a, bb, sc);
        }

        // ---- causal mask (before scale, as reference) + publish to LDS
#pragma unroll
        for (int r = 0; r < 8; ++r) {
            int row  = r + rh;
            int tcol = kvBase + w * 16 + cl;
            int srow = qRow0 + row;
            float val = (tcol <= srow) ? sc[r] * kscale : -1.0e30f;
            pT[row * PLD + w * 16 + cl] = val;
        }
        __syncthreads();

        // ---- online softmax: block-wide row max
        if (tid < 128) {
            int rr = tid & 15, part = tid >> 4;
            float mx = -3.0e38f;
#pragma unroll
            for (int c = 0; c < 16; ++c)
                mx = fmaxf(mx, pT[rr * PLD + part * 16 + c]);
            pmax[rr * 8 + part] = mx;
        }
        __syncthreads();
        if (tid < 16) {
            float mn = mR[tid];
#pragma unroll
            for (int p = 0; p < 8; ++p) mn = fmaxf(mn, pmax[tid * 8 + p]);
            fac[tid] = __expf(mR[tid] - mn);
            mR[tid]  = mn;
        }
        __syncthreads();
        // exp + bf16 probabilities (packed converts) + partial sums
        if (tid < 128) {
            int rr = tid & 15, part = tid >> 4;
            float mrow = mR[rr];
            float s = 0.f;
#pragma unroll
            for (int c = 0; c < 8; ++c) {
                int idx = rr * PLD + part * 16 + c * 2;
                float e0 = __expf(pT[idx + 0] - mrow);
                float e1 = __expf(pT[idx + 1] - mrow);
                *(bf16x2*)(pB + idx) = f32x2_to_bf16x2(e0, e1);
                s += e0 + e1;
            }
            psum[rr * 8 + part] = s;
        }
        __syncthreads();
        if (tid < 16) {
            float s = 0.f;
#pragma unroll
            for (int p = 0; p < 8; ++p) s += psum[tid * 8 + p];
            lR[tid] = lR[tid] * fac[tid] + s;
        }
        __syncthreads();

        // ---- rescale O and accumulate P . V  (wave w: V cols [w*128,+128))
        float fr[8];
#pragma unroll
        for (int r = 0; r < 8; ++r) fr[r] = fac[r + rh];
#pragma unroll
        for (int nt = 0; nt < 8; ++nt)
#pragma unroll
            for (int r = 0; r < 8; ++r) acc[nt][r] *= fr[r];

        bfrag pa[4];
#pragma unroll
        for (int ks = 0; ks < 4; ++ks)
            pa[ks] = load_a_frag(pB + ks * 32, PLD, lane);
#pragma unroll
        for (int nt = 0; nt < 8; ++nt) {
            // vT[d][s]: N (=d) rows, K (=kv position) contiguous
            const __bf16* vbase = vG + (size_t)(w * 128 + nt * 16) * S_DIM + kvBase;
#pragma unroll
            for (int ks = 0; ks < 4; ++ks) {
                bfrag bb = load_b_frag_t(vbase + ks * 32, S_DIM, lane);
                acc[nt] = wmma_bf16(pa[ks], bb, acc[nt]);
            }
        }
        __syncthreads();
    }

    // ---- finalize: O /= l, store bf16 row-major
    float invl[8];
#pragma unroll
    for (int r = 0; r < 8; ++r) invl[r] = 1.0f / lR[r + rh];
    __bf16* oG = O + ((size_t)bidx * S_DIM + qRow0) * D_DIM + w * 128;
#pragma unroll
    for (int nt = 0; nt < 8; ++nt)
#pragma unroll
        for (int r = 0; r < 8; ++r) {
            int row = r + rh;
            oG[(size_t)row * D_DIM + nt * 16 + cl] =
                f32_to_bf16(acc[nt][r] * invl[r]);
        }
}

// ---------------------------------------------------------------------------
// host-side launcher
// ---------------------------------------------------------------------------
extern "C" void kernel_launch(void* const* d_in, const int* in_sizes, int n_in,
                              void* d_out, int out_size, void* d_ws, size_t ws_size,
                              hipStream_t stream) {
    const float* x  = (const float*)d_in[0];
    const float* Wq = (const float*)d_in[1];
    const float* Wk = (const float*)d_in[2];
    const float* Wv = (const float*)d_in[3];
    const float* Wo = (const float*)d_in[4];
    float* out = (float*)d_out;

    char* ws = (char*)d_ws;
    size_t off = 0;
    const size_t tokBytes = (size_t)M_DIM * D_DIM * 2;  // 16 MB bf16
    const size_t wBytes   = (size_t)D_DIM * D_DIM * 2;  //  2 MB bf16

    __bf16* xb   = (__bf16*)(ws + off); off += tokBytes;
    __bf16* WqT  = (__bf16*)(ws + off); off += wBytes;   // [N][K]
    __bf16* WkT  = (__bf16*)(ws + off); off += wBytes;
    __bf16* WvT  = (__bf16*)(ws + off); off += wBytes;
    __bf16* WoT  = (__bf16*)(ws + off); off += wBytes;
    __bf16* qb   = (__bf16*)(ws + off); off += tokBytes; // [b*S][D] row-major
    __bf16* kb   = (__bf16*)(ws + off); off += tokBytes; // [b*S][D] row-major
    __bf16* vbT  = (__bf16*)(ws + off); off += tokBytes; // [b][D][S] transposed
    __bf16* ob   = (__bf16*)(ws + off); off += tokBytes; // [b*S][D] row-major

    // 1) conversions: x row-major, weights transposed
    const int nx = M_DIM * D_DIM;       // 8,388,608
    const int nw = D_DIM * D_DIM;       // 1,048,576
    cvt_f32_bf16<<<nx / 4 / 256, 256, 0, stream>>>(x, xb, nx);
    cvt_transpose_f32_bf16<<<nw / 256, 256, 0, stream>>>(Wq, WqT, D_DIM, D_DIM);
    cvt_transpose_f32_bf16<<<nw / 256, 256, 0, stream>>>(Wk, WkT, D_DIM, D_DIM);
    cvt_transpose_f32_bf16<<<nw / 256, 256, 0, stream>>>(Wv, WvT, D_DIM, D_DIM);
    cvt_transpose_f32_bf16<<<nw / 256, 256, 0, stream>>>(Wo, WoT, D_DIM, D_DIM);

    // 2) projections  (8192x1024 @ 1024x1024)
    dim3 gg(D_DIM / 128, M_DIM / 128);  // (8, 64)
    dim3 bb(256);
    gemm_bf16_bt<0><<<gg, bb, 0, stream>>>(xb, WqT, qb,  M_DIM, D_DIM, D_DIM);
    gemm_bf16_bt<0><<<gg, bb, 0, stream>>>(xb, WkT, kb,  M_DIM, D_DIM, D_DIM);
    gemm_bf16_bt<1><<<gg, bb, 0, stream>>>(xb, WvT, vbT, M_DIM, D_DIM, D_DIM);

    // 3) causal flash attention -> ob (bf16)
    dim3 gf(S_DIM / 16, B_DIM);         // (128, 4)
    flash_attn<<<gf, bb, 0, stream>>>(qb, kb, vbT, ob);

    // 4) output projection -> fp32 d_out
    gemm_bf16_bt<2><<<gg, bb, 0, stream>>>(ob, WoT, out, M_DIM, D_DIM, D_DIM);
}
